// torch_GRL_44830868635693
// MI455X (gfx1250) — matrix operand order repaired and checked by reading, not compiled, not run on hardware
//
#include <hip/hip_runtime.h>

#define NN   16384
#define MAXE 128
#define ASTR 40   // LDS row stride in halves: 80B rows -> 16B aligned, bank-stride 20

typedef _Float16 v8h  __attribute__((ext_vector_type(8)));
typedef _Float16 v16h __attribute__((ext_vector_type(16)));
typedef float    v8f  __attribute__((ext_vector_type(8)));
typedef int      v4i  __attribute__((ext_vector_type(4)));

#if __has_builtin(__builtin_amdgcn_global_load_async_to_lds_b128)
#define SCAN_ASYNC 1
#else
#define SCAN_ASYNC 0
#endif

#if SCAN_ASYNC
#if __has_builtin(__builtin_amdgcn_s_wait_asynccnt)
#define WAIT_ASYNC(n) __builtin_amdgcn_s_wait_asynccnt(n)
#else
#define WAIT_ASYNC(n) asm volatile("s_wait_asynccnt %0" ::"n"(n) : "memory")
#endif
// Param 1 of the builtin is 'int4 AS(1)*' (clang prints AS1 as __device__);
// mirror the same element type for the LDS-side pointer (AS3).
#define GPTR(p) ((__attribute__((address_space(1))) v4i*)(p))
#define LPTR(p) ((__attribute__((address_space(3))) v4i*)(p))
#endif

static __device__ inline v8f vzero8f() {
  v8f z = {0.f, 0.f, 0.f, 0.f, 0.f, 0.f, 0.f, 0.f};
  return z;
}

static __device__ inline v8f wmma32(v16h a, v16h b, v8f c) {
  // D = A(16x32 f16) x B(32x16 f16) + C(16x16 f32)
  return __builtin_amdgcn_wmma_f32_16x16x32_f16(false, a, false, b, (short)0, c,
                                                false, false);
}

// A fragment (16x32) from a global f32 row-major matrix (leading dim ld).
// ISA layout: lane l holds row m=l&15; halves 0..7 = K=h*8+i, 8..15 = K=16+h*8+i (h=l>>4).
static __device__ inline v16h afrag_g(const float* __restrict__ base, int ld, int lane) {
  int m = lane & 15, h = lane >> 4;
  const float* p = base + (size_t)m * ld + h * 8;
  v16h a;
#pragma unroll
  for (int i = 0; i < 8; ++i) {
    a[i]     = (_Float16)p[i];
    a[i + 8] = (_Float16)p[16 + i];
  }
  return a;
}

// A fragment from an LDS f16 tile [16][ASTR].
static __device__ inline v16h afrag_l(const _Float16* t, int lane) {
  int m = lane & 15, h = lane >> 4;
  v8h lo = *(const v8h*)(t + m * ASTR + h * 8);
  v8h hi = *(const v8h*)(t + m * ASTR + 16 + h * 8);
  v16h a;
#pragma unroll
  for (int i = 0; i < 8; ++i) { a[i] = lo[i]; a[i + 8] = hi[i]; }
  return a;
}

// B fragment (32x16) from LDS f16 weight tile [K][ASTR]: lane holds col N=nbase+(lane&15);
// halves 0..7 = K=kbase+h*8+i, 8..15 = K=kbase+16+h*8+i (h=lane>>4).
static __device__ inline v16h bfrag_l(const _Float16* w, int kbase, int nbase, int lane) {
  int n = lane & 15, h = lane >> 4;
  const _Float16* p = w + (kbase + h * 8) * ASTR + nbase + n;
  v16h b;
#pragma unroll
  for (int i = 0; i < 8; ++i) { b[i] = p[i * ASTR]; b[i + 8] = p[(16 + i) * ASTR]; }
  return b;
}

// Write C tile (16x16 f32) into LDS act tile as f16 with bias + relu.
// C layout: lane l -> col n=l&15 (+16*tcol), rows (l>>4)*8 + v.
static __device__ inline void stage_relu(_Float16* t, v8f c, int tcol,
                                         const float* __restrict__ bias, int lane) {
  int n  = (lane & 15) + tcol * 16;
  int mb = (lane >> 4) * 8;
  float b = bias[n];
#pragma unroll
  for (int v = 0; v < 8; ++v) {
    float x = fmaxf(c[v] + b, 0.f);
    t[(mb + v) * ASTR + n] = (_Float16)x;
  }
}

// ---------------------------------------------------------------------------
// Pass 1: stream A once; emit ns[row]=rsqrt(deg+1) and compact u16 edge lists.
// One block (512 threads) per row (64 KB). If the toolchain exposes the CDNA5
// async-LDS DMA builtins, double-buffer 8 KB chunks global->LDS with
// GLOBAL_LOAD_ASYNC_TO_LDS_B128 (ASYNCcnt); else fall back to plain b128 loads.
// ---------------------------------------------------------------------------
__global__ __launch_bounds__(512) void scan_kernel(const float* __restrict__ Adj,
                                                   unsigned short* __restrict__ elist,
                                                   int* __restrict__ ecnt,
                                                   float* __restrict__ ns) {
  __shared__ int cnt;
  __shared__ unsigned short lst[MAXE];
#if SCAN_ASYNC
  __shared__ float buf[2][2048];  // 2 x 8 KB double buffer
#endif
  int row = blockIdx.x, tid = threadIdx.x;
  if (tid == 0) cnt = 0;
  __syncthreads();

#if SCAN_ASYNC
  const float* rowp = Adj + (size_t)row * NN;
  // prime chunk 0: each lane copies 16 B (512 threads -> 8 KB)
  __builtin_amdgcn_global_load_async_to_lds_b128(GPTR(rowp + tid * 4),
                                                 LPTR(&buf[0][tid * 4]), 0, 0);
#pragma unroll
  for (int c = 0; c < 8; ++c) {
    if (c < 7) {
      __builtin_amdgcn_global_load_async_to_lds_b128(
          GPTR(rowp + (c + 1) * 2048 + tid * 4), LPTR(&buf[(c + 1) & 1][tid * 4]), 0, 0);
      WAIT_ASYNC(1);  // own copy of chunk c has landed
    } else {
      WAIT_ASYNC(0);
    }
    __syncthreads();  // all 16 waves' copies of chunk c visible
    float4 v = *(const float4*)&buf[c & 1][tid * 4];
    int jb = c * 2048 + tid * 4;
    if (v.x != 0.f) { int s = atomicAdd(&cnt, 1); if (s < MAXE) lst[s] = (unsigned short)(jb + 0); }
    if (v.y != 0.f) { int s = atomicAdd(&cnt, 1); if (s < MAXE) lst[s] = (unsigned short)(jb + 1); }
    if (v.z != 0.f) { int s = atomicAdd(&cnt, 1); if (s < MAXE) lst[s] = (unsigned short)(jb + 2); }
    if (v.w != 0.f) { int s = atomicAdd(&cnt, 1); if (s < MAXE) lst[s] = (unsigned short)(jb + 3); }
    __syncthreads();  // reads done before this buffer is overwritten (chunk c+2)
  }
#else
  const float4* p = (const float4*)(Adj + (size_t)row * NN) + tid * 8;
  __builtin_prefetch(p, 0, 0);  // global_prefetch_b8
  __builtin_prefetch(p + 4, 0, 0);
#pragma unroll
  for (int q = 0; q < 8; ++q) {
    float4 v = p[q];
    int jb = tid * 32 + q * 4;
    if (v.x != 0.f) { int s = atomicAdd(&cnt, 1); if (s < MAXE) lst[s] = (unsigned short)(jb + 0); }
    if (v.y != 0.f) { int s = atomicAdd(&cnt, 1); if (s < MAXE) lst[s] = (unsigned short)(jb + 1); }
    if (v.z != 0.f) { int s = atomicAdd(&cnt, 1); if (s < MAXE) lst[s] = (unsigned short)(jb + 2); }
    if (v.w != 0.f) { int s = atomicAdd(&cnt, 1); if (s < MAXE) lst[s] = (unsigned short)(jb + 3); }
  }
  __syncthreads();
#endif

  int c = cnt;
  if (tid == 0) {
    ns[row]   = rsqrtf((float)c + 1.0f);
    ecnt[row] = (c < MAXE) ? c : MAXE;
  }
  int cl = (c < MAXE) ? c : MAXE;
  for (int i = tid; i < cl; i += 512) elist[(size_t)row * MAXE + i] = lst[i];
}

// ---------------------------------------------------------------------------
// Pass 2: encoder. X2 = relu(relu(Xin@W1+b1)@W2+b2); scaled = ns * (X2@Wg).
// 128 threads = 4 waves; each wave owns a 16-row strip (WMMA).
// ---------------------------------------------------------------------------
__global__ __launch_bounds__(128) void encoder_kernel(
    const float* __restrict__ Xin,
    const float* __restrict__ W1, const float* __restrict__ b1,
    const float* __restrict__ W2, const float* __restrict__ b2,
    const float* __restrict__ Wg, const float* __restrict__ ns,
    float* __restrict__ X2g, float* __restrict__ scl) {
  __shared__ _Float16 sW1[128 * ASTR];
  __shared__ _Float16 sW2[32 * ASTR];
  __shared__ _Float16 sWg[32 * ASTR];
  __shared__ _Float16 sAct[4][16 * ASTR];
  int tid = threadIdx.x;
  for (int i = tid; i < 128 * 32; i += 128) sW1[(i >> 5) * ASTR + (i & 31)] = (_Float16)W1[i];
  for (int i = tid; i < 32 * 32; i += 128) {
    sW2[(i >> 5) * ASTR + (i & 31)] = (_Float16)W2[i];
    sWg[(i >> 5) * ASTR + (i & 31)] = (_Float16)Wg[i];
  }
  __syncthreads();
  int wave = tid >> 5, lane = tid & 31;
  int r0 = (blockIdx.x * 4 + wave) * 16;
  _Float16* act = sAct[wave];

  // layer 1: (16x128) @ (128x32), K-loop of 4
  v8f c0 = vzero8f(), c1 = vzero8f();
#pragma unroll
  for (int s = 0; s < 4; ++s) {
    v16h a = afrag_g(Xin + (size_t)r0 * 128 + s * 32, 128, lane);
    c0 = wmma32(a, bfrag_l(sW1, s * 32, 0, lane), c0);
    c1 = wmma32(a, bfrag_l(sW1, s * 32, 16, lane), c1);
  }
  stage_relu(act, c0, 0, b1, lane);
  stage_relu(act, c1, 1, b1, lane);
  __syncthreads();
  v16h aX1 = afrag_l(act, lane);
  __syncthreads();

  // layer 2: (16x32) @ (32x32); write X2 (f32) and stage f16 copy
  c0 = vzero8f(); c1 = vzero8f();
  c0 = wmma32(aX1, bfrag_l(sW2, 0, 0, lane), c0);
  c1 = wmma32(aX1, bfrag_l(sW2, 0, 16, lane), c1);
  {
    int n0 = lane & 15, mb = (lane >> 4) * 8;
    float bb0 = b2[n0], bb1 = b2[n0 + 16];
#pragma unroll
    for (int v = 0; v < 8; ++v) {
      int row = r0 + mb + v;
      float x0 = fmaxf(c0[v] + bb0, 0.f);
      float x1 = fmaxf(c1[v] + bb1, 0.f);
      X2g[(size_t)row * 32 + n0]      = x0;
      X2g[(size_t)row * 32 + n0 + 16] = x1;
      act[(mb + v) * ASTR + n0]      = (_Float16)x0;
      act[(mb + v) * ASTR + n0 + 16] = (_Float16)x1;
    }
  }
  __syncthreads();
  v16h aX2 = afrag_l(act, lane);

  // Xw = X2 @ Wg; scaled = ns[row] * Xw (no bias/relu here)
  c0 = vzero8f(); c1 = vzero8f();
  c0 = wmma32(aX2, bfrag_l(sWg, 0, 0, lane), c0);
  c1 = wmma32(aX2, bfrag_l(sWg, 0, 16, lane), c1);
  {
    int n0 = lane & 15, mb = (lane >> 4) * 8;
#pragma unroll
    for (int v = 0; v < 8; ++v) {
      int row = r0 + mb + v;
      float s = ns[row];
      scl[(size_t)row * 32 + n0]      = c0[v] * s;
      scl[(size_t)row * 32 + n0 + 16] = c1[v] * s;
    }
  }
}

// ---------------------------------------------------------------------------
// Pass 3: aggregation. X_graph = relu(ns*(B@scaled + scaled) + bg).
// One wave per row, lane = feature column; 'scaled' (2MB) is L2-resident.
// ---------------------------------------------------------------------------
__global__ __launch_bounds__(256) void spmm_kernel(
    const unsigned short* __restrict__ elist, const int* __restrict__ ecnt,
    const float* __restrict__ ns, const float* __restrict__ scl,
    const float* __restrict__ bg, float* __restrict__ Xg) {
  int gid = blockIdx.x * 256 + threadIdx.x;
  int row = gid >> 5, lane = gid & 31;
  float acc = scl[(size_t)row * 32 + lane];  // self-loop term
  int cnt = ecnt[row];
  const unsigned short* l = elist + (size_t)row * MAXE;
  for (int e = 0; e < cnt; ++e) {
    int j = l[e];
    acc += scl[(size_t)j * 32 + lane];
  }
  float v = ns[row] * acc + bg[lane];
  Xg[(size_t)row * 32 + lane] = fmaxf(v, 0.f);
}

// ---------------------------------------------------------------------------
// Pass 4: head. Xg2=relu(Xg@Wgd+bgd); concat(Xg2,X2)@Wp1 -> relu -> @Wp2
// -> relu -> @Wpo + bpo, * RL. WMMA per 16-row strip.
// ---------------------------------------------------------------------------
__global__ __launch_bounds__(128) void head_kernel(
    const float* __restrict__ Xg, const float* __restrict__ X2g,
    const float* __restrict__ Wgd, const float* __restrict__ bgd,
    const float* __restrict__ Wp1, const float* __restrict__ bp1,
    const float* __restrict__ Wp2, const float* __restrict__ bp2,
    const float* __restrict__ Wpo, const float* __restrict__ bpo,
    const float* __restrict__ RL, float* __restrict__ out) {
  __shared__ _Float16 sWgd[32 * ASTR];
  __shared__ _Float16 sWp1[64 * ASTR];
  __shared__ _Float16 sWp2[32 * ASTR];
  __shared__ _Float16 sWpo[32 * ASTR];
  __shared__ _Float16 sAct[4][16 * ASTR];
  int tid = threadIdx.x;
  for (int i = tid; i < 32 * 32; i += 128) {
    sWgd[(i >> 5) * ASTR + (i & 31)] = (_Float16)Wgd[i];
    sWp2[(i >> 5) * ASTR + (i & 31)] = (_Float16)Wp2[i];
  }
  for (int i = tid; i < 64 * 32; i += 128) sWp1[(i >> 5) * ASTR + (i & 31)] = (_Float16)Wp1[i];
  for (int i = tid; i < 32 * 16; i += 128) {
    int k = i >> 4, n = i & 15;
    sWpo[k * ASTR + n] = (n < 8) ? (_Float16)Wpo[k * 8 + n] : (_Float16)0.f;  // pad N to 16
  }
  __syncthreads();
  int wave = tid >> 5, lane = tid & 31;
  int r0 = (blockIdx.x * 4 + wave) * 16;
  _Float16* act = sAct[wave];

  // Xg2 = relu(Xg @ Wgd + bgd)
  v16h aXg = afrag_g(Xg + (size_t)r0 * 32, 32, lane);
  v8f c0 = vzero8f(), c1 = vzero8f();
  c0 = wmma32(aXg, bfrag_l(sWgd, 0, 0, lane), c0);
  c1 = wmma32(aXg, bfrag_l(sWgd, 0, 16, lane), c1);
  stage_relu(act, c0, 0, bgd, lane);
  stage_relu(act, c1, 1, bgd, lane);
  __syncthreads();
  v16h aG  = afrag_l(act, lane);
  v16h aX2 = afrag_g(X2g + (size_t)r0 * 32, 32, lane);  // second half of concat
  __syncthreads();

  // concat(Xg2, X2) @ Wp1 : K=64 split into two WMMA K-steps
  c0 = vzero8f(); c1 = vzero8f();
  c0 = wmma32(aG, bfrag_l(sWp1, 0, 0, lane), c0);
  c1 = wmma32(aG, bfrag_l(sWp1, 0, 16, lane), c1);
  c0 = wmma32(aX2, bfrag_l(sWp1, 32, 0, lane), c0);
  c1 = wmma32(aX2, bfrag_l(sWp1, 32, 16, lane), c1);
  stage_relu(act, c0, 0, bp1, lane);
  stage_relu(act, c1, 1, bp1, lane);
  __syncthreads();
  v16h aP = afrag_l(act, lane);
  __syncthreads();

  c0 = vzero8f(); c1 = vzero8f();
  c0 = wmma32(aP, bfrag_l(sWp2, 0, 0, lane), c0);
  c1 = wmma32(aP, bfrag_l(sWp2, 0, 16, lane), c1);
  stage_relu(act, c0, 0, bp2, lane);
  stage_relu(act, c1, 1, bp2, lane);
  __syncthreads();
  v16h aQ = afrag_l(act, lane);

  // output layer: (16x32) @ (32x8 padded to 16), then bias + RL mask
  v8f o = vzero8f();
  o = wmma32(aQ, bfrag_l(sWpo, 0, 0, lane), o);
  int n = lane & 15, mb = (lane >> 4) * 8;
  if (n < 8) {
    float bo = bpo[n];
#pragma unroll
    for (int v = 0; v < 8; ++v) {
      int row = r0 + mb + v;
      out[(size_t)row * 8 + n] = (o[v] + bo) * RL[row];
    }
  }
}

// ---------------------------------------------------------------------------
extern "C" void kernel_launch(void* const* d_in, const int* in_sizes, int n_in,
                              void* d_out, int out_size, void* d_ws, size_t ws_size,
                              hipStream_t stream) {
  (void)in_sizes; (void)n_in; (void)out_size; (void)ws_size;
  const float* Xin = (const float*)d_in[0];
  const float* Adj = (const float*)d_in[1];
  const float* RL  = (const float*)d_in[2];
  const float* W1  = (const float*)d_in[3];  const float* b1  = (const float*)d_in[4];
  const float* W2  = (const float*)d_in[5];  const float* b2  = (const float*)d_in[6];
  const float* Wg  = (const float*)d_in[7];  const float* bg  = (const float*)d_in[8];
  const float* Wgd = (const float*)d_in[9];  const float* bgd = (const float*)d_in[10];
  const float* Wp1 = (const float*)d_in[11]; const float* bp1 = (const float*)d_in[12];
  const float* Wp2 = (const float*)d_in[13]; const float* bp2 = (const float*)d_in[14];
  const float* Wpo = (const float*)d_in[15]; const float* bpo = (const float*)d_in[16];

  char* ws = (char*)d_ws;
  float*          ns    = (float*)(ws + 0);                          //  64 KB
  int*            ecnt  = (int*)(ws + (size_t)65536);                //  64 KB
  unsigned short* elist = (unsigned short*)(ws + (size_t)131072);    //   4 MB
  float*          X2g   = (float*)(ws + (size_t)4325376);            //   2 MB
  float*          scl   = (float*)(ws + (size_t)6422528);            //   2 MB
  float*          Xgr   = (float*)(ws + (size_t)8519680);            //   2 MB

  scan_kernel<<<NN, 512, 0, stream>>>(Adj, elist, ecnt, ns);
  encoder_kernel<<<NN / 64, 128, 0, stream>>>(Xin, W1, b1, W2, b2, Wg, ns, X2g, scl);
  spmm_kernel<<<(NN * 32) / 256, 256, 0, stream>>>(elist, ecnt, ns, scl, bg, Xgr);
  head_kernel<<<NN / 64, 128, 0, stream>>>(Xgr, X2g, Wgd, bgd, Wp1, bp1, Wp2, bp2,
                                           Wpo, bpo, RL, (float*)d_out);
}
